// MultiHeadAttention_29068338659532
// MI455X (gfx1250) — matrix operand order repaired and checked by reading, not compiled, not run on hardware
//
#include <hip/hip_runtime.h>

#define D_MODEL 1024
#define N_HEADS 16
#define DEPTH   64
#define BATCH   4
#define SEQ     2048

typedef __attribute__((ext_vector_type(16))) __bf16          v16bf;
typedef __attribute__((ext_vector_type(8)))  float           v8f;
typedef __attribute__((ext_vector_type(4)))  float           v4f;
typedef __attribute__((ext_vector_type(4)))  unsigned short  v4u16;

// ---------------- bf16 helpers ----------------
static __device__ __forceinline__ unsigned short f32_to_bf16(float f) {
  unsigned int u = __float_as_uint(f);
  unsigned int r = 0x7FFFu + ((u >> 16) & 1u);   // round-to-nearest-even
  return (unsigned short)((u + r) >> 16);
}
static __device__ __forceinline__ __bf16 asbf(unsigned short h) {
  return __builtin_bit_cast(__bf16, h);
}

// ---------------- WMMA fragment loaders (CDNA5 wave32 layouts) ----------------
// A matrix 16x32 (MxK), 16-bit: lane m = lane&15; lanes 0-15 hold K={0..7,16..23},
// lanes 16-31 hold K={8..15,24..31}. Per lane: two contiguous 8-half runs -> 2x b128.
static __device__ __forceinline__ v16bf load_frag_a(const unsigned short* src, int lda) {
  const int lane = threadIdx.x & 31;
  const int m  = lane & 15;
  const int kb = (lane >> 4) << 3;
  const unsigned short* p = src + m * lda + kb;
  v16bf a;
#pragma unroll
  for (int i = 0; i < 8; ++i) a[i]     = asbf(p[i]);
#pragma unroll
  for (int i = 0; i < 8; ++i) a[8 + i] = asbf(p[16 + i]);
  return a;
}
// B matrix 32x16 (KxN) from a TRANSPOSED [n][k] tile: lane n = lane&15;
// lanes 0-15 hold K={0..7,16..23}, lanes 16-31 hold K={8..15,24..31} -> two 8-half runs per lane.
static __device__ __forceinline__ v16bf load_frag_bt(const unsigned short* src, int ldb) {
  const int lane = threadIdx.x & 31;
  const int nn = lane & 15;
  const int k0 = (lane >> 4) << 3;
  const unsigned short* p = src + nn * ldb + k0;
  v16bf bv;
#pragma unroll
  for (int j = 0; j < 8; ++j) bv[j] = asbf(p[j]);
#pragma unroll
  for (int j = 0; j < 8; ++j) bv[8 + j] = asbf(p[16 + j]);
  return bv;
}
static __device__ __forceinline__ v8f wmma_bf16(v16bf a, v16bf b, v8f c) {
  return __builtin_amdgcn_wmma_f32_16x16x32_bf16(false, a, false, b, (short)0, c, false, false);
}

// ---------------- cooperative LDS staging (128 threads) ----------------
// 64x32 row-major halves tile, b128 path both sides
static __device__ __forceinline__ void stage_rm_64x32(unsigned short* lds, int ldl,
                                                      const unsigned short* g, int ldg) {
#pragma unroll
  for (int j = 0; j < 2; ++j) {
    int q = j * 128 + (int)threadIdx.x;
    int row = q >> 2, coff = (q & 3) * 8;
    *(uint4*)(lds + row * ldl + coff) =
        *(const uint4*)(g + (size_t)row * ldg + coff);
  }
}
// 32x64 row-major halves tile, b128 path both sides
static __device__ __forceinline__ void stage_rm_32x64(unsigned short* lds, int ldl,
                                                      const unsigned short* g, int ldg) {
#pragma unroll
  for (int j = 0; j < 2; ++j) {
    int q = j * 128 + (int)threadIdx.x;
    int row = q >> 3, coff = (q & 7) * 8;
    *(uint4*)(lds + row * ldl + coff) =
        *(const uint4*)(g + (size_t)row * ldg + coff);
  }
}
// 32x64 global [k][n] -> LDS transposed [n][k] (scatter stores; fragment reads become b128)
static __device__ __forceinline__ void stage_tr_32x64(unsigned short* lds, int ldl,
                                                      const unsigned short* g, int ldg) {
#pragma unroll
  for (int j = 0; j < 2; ++j) {
    int q = j * 128 + (int)threadIdx.x;
    int k = q >> 3, noff = (q & 7) * 8;
    union { uint4 v; unsigned short h[8]; } u;
    u.v = *(const uint4*)(g + (size_t)k * ldg + noff);
#pragma unroll
    for (int e = 0; e < 8; ++e) lds[(noff + e) * ldl + k] = u.h[e];
  }
}

// ---------------- fp32 -> bf16 conversion (vectorized, NT loads) ----------------
__global__ void cvt_f32_bf16_kernel(const float* __restrict__ in,
                                    unsigned short* __restrict__ out, int n4) {
  int i = blockIdx.x * blockDim.x + threadIdx.x;
  if (i < n4) {
    v4f f = __builtin_nontemporal_load((const v4f*)in + i);
    v4u16 o;
    o[0] = f32_to_bf16(f[0]); o[1] = f32_to_bf16(f[1]);
    o[2] = f32_to_bf16(f[2]); o[3] = f32_to_bf16(f[3]);
    *((v4u16*)out + i) = o;
  }
}

#define XSTR 40   // 64x32 A tile stride (halves); 80B rows -> 16B aligned
#define BSTR 40   // 64x32 transposed-B tile stride
#define KSTR 72   // 32x64 K block stride (144B rows, 16B aligned)
#define VSTR 40   // 64x32 transposed-V tile stride
#define PSTR 40   // 16x32 P tile stride

// ---------------- QKV projection: Q/K/V[b,h,s,e] = x @ W(h) + bias ----------------
__global__ void __launch_bounds__(128) qkv_proj_kernel(
    const unsigned short* __restrict__ xb,
    const unsigned short* __restrict__ Wqb, const unsigned short* __restrict__ Wkb,
    const unsigned short* __restrict__ Wvb,
    const float* __restrict__ bq, const float* __restrict__ bk, const float* __restrict__ bv,
    unsigned short* __restrict__ Q, unsigned short* __restrict__ K,
    unsigned short* __restrict__ V) {
  const int h   = blockIdx.y;
  const int m0  = blockIdx.x * 64;           // row tile over B*S
  const int sel = blockIdx.z;
  const unsigned short* W = (sel == 0) ? Wqb : (sel == 1) ? Wkb : Wvb;
  const float* bias       = (sel == 0) ? bq  : (sel == 1) ? bk  : bv;
  unsigned short* Out     = (sel == 0) ? Q   : (sel == 1) ? K   : V;

  __shared__ unsigned short xs[64 * XSTR];
  __shared__ unsigned short wt[64 * BSTR];   // W tile, transposed [n][k]

  const int tid = threadIdx.x, lane = tid & 31, w = tid >> 5;
  const unsigned short* Wh = W + (size_t)h * D_MODEL * DEPTH;

  v8f acc[4] = {};
  for (int kk = 0; kk < D_MODEL; kk += 32) {
    __syncthreads();
    stage_rm_64x32(xs, XSTR, xb + (size_t)m0 * D_MODEL + kk, D_MODEL);
    stage_tr_32x64(wt, BSTR, Wh + (size_t)kk * DEPTH, DEPTH);
    __syncthreads();
    v16bf a = load_frag_a(xs + (w * 16) * XSTR, XSTR);
#pragma unroll
    for (int t = 0; t < 4; ++t) {
      v16bf bfr = load_frag_bt(wt + (t * 16) * BSTR, BSTR);
      acc[t] = wmma_bf16(a, bfr, acc[t]);
    }
  }
  const int n = lane & 15, half = lane >> 4;
  const float* bh = bias + h * DEPTH;
  const int mbase = m0 + w * 16 + half * 8;
#pragma unroll
  for (int t = 0; t < 4; ++t)
#pragma unroll
    for (int r = 0; r < 8; ++r) {
      float v = acc[t][r] + bh[t * 16 + n];
      int mg = mbase + r;
      int bb = mg >> 11, s = mg & (SEQ - 1);     // SEQ = 2048
      Out[(((size_t)bb * N_HEADS + h) * SEQ + s) * DEPTH + t * 16 + n] = f32_to_bf16(v);
    }
}

// ---------------- Flash attention per (b,h,64-q-tile) ----------------
__global__ void __launch_bounds__(128) attn_kernel(
    const unsigned short* __restrict__ Q, const unsigned short* __restrict__ K,
    const unsigned short* __restrict__ V, unsigned short* __restrict__ Z) {
  const int h = blockIdx.y, b = blockIdx.z;
  const int q0 = blockIdx.x * 64;
  const int tid = threadIdx.x, lane = tid & 31, w = tid >> 5;
  const int n = lane & 15, half = lane >> 4;

  __shared__ unsigned short Kb[32 * KSTR];   // [kpos][e], row-major
  __shared__ unsigned short Vt[64 * VSTR];   // V transposed: [e][kpos]
  __shared__ unsigned short Pl[4 * 16 * PSTR];

  const size_t bh = ((size_t)b * N_HEADS + h) * SEQ;
  const unsigned short* Qg = Q + (bh + q0 + (size_t)w * 16) * DEPTH;

  v16bf aQ[2];
#pragma unroll
  for (int ec = 0; ec < 2; ++ec) aQ[ec] = load_frag_a(Qg + ec * 32, DEPTH);

  v8f acc[4] = {};
  float mi[8], li[8];
#pragma unroll
  for (int r = 0; r < 8; ++r) { mi[r] = -3.0e38f; li[r] = 0.f; }
  unsigned short* Pw = Pl + w * 16 * PSTR;

  for (int kb0 = 0; kb0 < SEQ; kb0 += 32) {
    __syncthreads();
    stage_rm_32x64(Kb, KSTR, K + (bh + kb0) * DEPTH, DEPTH);
    stage_tr_32x64(Vt, VSTR, V + (bh + kb0) * DEPTH, DEPTH);
    if (kb0 + 32 < SEQ) {   // prefetch next K/V block (whole block covered by 128 threads)
      __builtin_prefetch(K + (bh + kb0 + 32) * DEPTH + tid * 16, 0, 0);
      __builtin_prefetch(V + (bh + kb0 + 32) * DEPTH + tid * 16, 0, 0);
    }
    __syncthreads();

    // scores: S = Q @ K^T  (two 16-col tiles); B(e,kpos) = Kb[kpos][e] -> contiguous
    v8f s0 = {}, s1 = {};
#pragma unroll
    for (int ec = 0; ec < 2; ++ec) {
      v16bf b0 = load_frag_bt(Kb + ec * 32, KSTR);
      s0 = wmma_bf16(aQ[ec], b0, s0);
      v16bf b1 = load_frag_bt(Kb + 16 * KSTR + ec * 32, KSTR);
      s1 = wmma_bf16(aQ[ec], b1, s1);
    }

    // online softmax over 32 new columns
    const float sc = 0.125f;                     // 1/sqrt(64)
    float mx[8], al[8], rs[8];
    v8f p0, p1;
#pragma unroll
    for (int r = 0; r < 8; ++r) {
      s0[r] *= sc; s1[r] *= sc;
      mx[r] = fmaxf(s0[r], s1[r]);
    }
#pragma unroll
    for (int msk = 1; msk < 16; msk <<= 1)
#pragma unroll
      for (int r = 0; r < 8; ++r)
        mx[r] = fmaxf(mx[r], __shfl_xor(mx[r], msk, 32));
#pragma unroll
    for (int r = 0; r < 8; ++r) {
      float mnew = fmaxf(mi[r], mx[r]);
      al[r] = __expf(mi[r] - mnew);
      float e0 = __expf(s0[r] - mnew), e1 = __expf(s1[r] - mnew);
      p0[r] = e0; p1[r] = e1;
      rs[r] = e0 + e1;
      mi[r] = mnew;
    }
#pragma unroll
    for (int msk = 1; msk < 16; msk <<= 1)
#pragma unroll
      for (int r = 0; r < 8; ++r)
        rs[r] += __shfl_xor(rs[r], msk, 32);
#pragma unroll
    for (int r = 0; r < 8; ++r) li[r] = li[r] * al[r] + rs[r];
#pragma unroll
    for (int t = 0; t < 4; ++t)
#pragma unroll
      for (int r = 0; r < 8; ++r) acc[t][r] *= al[r];

    // C-layout P -> LDS -> A-layout (transpose through LDS)
#pragma unroll
    for (int r = 0; r < 8; ++r) {
      int row = r + half * 8;
      Pw[row * PSTR + n]      = f32_to_bf16(p0[r]);
      Pw[row * PSTR + 16 + n] = f32_to_bf16(p1[r]);
    }
    __syncthreads();

    v16bf aP = load_frag_a(Pw, PSTR);
#pragma unroll
    for (int t = 0; t < 4; ++t) {
      v16bf bV = load_frag_bt(Vt + (t * 16) * VSTR, VSTR);  // B(k,e) = Vt[e][k]
      acc[t] = wmma_bf16(aP, bV, acc[t]);
    }
  }

#pragma unroll
  for (int r = 0; r < 8; ++r) li[r] = 1.0f / li[r];
  // Z layout: [B, S, H*DEPTH] (head-major concat matches reference transpose+reshape)
  size_t rowbase = ((size_t)b * SEQ + q0 + w * 16 + half * 8) * D_MODEL + (size_t)h * DEPTH;
#pragma unroll
  for (int t = 0; t < 4; ++t)
#pragma unroll
    for (int r = 0; r < 8; ++r)
      Z[rowbase + (size_t)r * D_MODEL + t * 16 + n] = f32_to_bf16(acc[t][r] * li[r]);
}

// ---------------- Output projection: out = Z @ Wo + bo (fp32 out, NT stores) ----------------
__global__ void __launch_bounds__(128) oproj_kernel(
    const unsigned short* __restrict__ Zb, const unsigned short* __restrict__ Wob,
    const float* __restrict__ bo, float* __restrict__ out) {
  const int m0 = blockIdx.x * 64;
  const int n0 = blockIdx.y * 64;
  const int tid = threadIdx.x, lane = tid & 31, w = tid >> 5;
  __shared__ unsigned short zs[64 * XSTR];
  __shared__ unsigned short wt[64 * BSTR];   // Wo tile transposed [n][k]

  v8f acc[4] = {};
  for (int kk = 0; kk < D_MODEL; kk += 32) {
    __syncthreads();
    stage_rm_64x32(zs, XSTR, Zb + (size_t)m0 * D_MODEL + kk, D_MODEL);
    stage_tr_32x64(wt, BSTR, Wob + (size_t)kk * D_MODEL + n0, D_MODEL);
    __syncthreads();
    v16bf a = load_frag_a(zs + (w * 16) * XSTR, XSTR);
#pragma unroll
    for (int t = 0; t < 4; ++t) {
      v16bf bb = load_frag_bt(wt + (t * 16) * BSTR, BSTR);
      acc[t] = wmma_bf16(a, bb, acc[t]);
    }
  }
  const int n = lane & 15, half = lane >> 4;
  size_t rowbase = (size_t)(m0 + w * 16 + half * 8) * D_MODEL + n0;
#pragma unroll
  for (int t = 0; t < 4; ++t)
#pragma unroll
    for (int r = 0; r < 8; ++r)
      __builtin_nontemporal_store(acc[t][r] + bo[n0 + t * 16 + n],
                                  &out[rowbase + (size_t)r * D_MODEL + t * 16 + n]);
}

// ---------------- host launcher ----------------
extern "C" void kernel_launch(void* const* d_in, const int* in_sizes, int n_in,
                              void* d_out, int out_size, void* d_ws, size_t ws_size,
                              hipStream_t stream) {
  const float* x  = (const float*)d_in[0];
  const float* Wq = (const float*)d_in[1];
  const float* bq = (const float*)d_in[2];
  const float* Wk = (const float*)d_in[3];
  const float* bk = (const float*)d_in[4];
  const float* Wv = (const float*)d_in[5];
  const float* bv = (const float*)d_in[6];
  const float* Wo = (const float*)d_in[7];
  const float* bo = (const float*)d_in[8];
  float* out = (float*)d_out;

  const size_t NX = (size_t)BATCH * SEQ * D_MODEL;           // 8,388,608
  const size_t NW = (size_t)N_HEADS * D_MODEL * DEPTH;       // 1,048,576
  const size_t NO = (size_t)D_MODEL * D_MODEL;               // 1,048,576
  const size_t NQ = (size_t)BATCH * N_HEADS * SEQ * DEPTH;   // 8,388,608

  char* ws = (char*)d_ws;
  size_t off = 0;
  auto alloc = [&](size_t bytes) -> unsigned short* {
    unsigned short* p = (unsigned short*)(ws + off);
    off += (bytes + 255) & ~(size_t)255;
    return p;
  };
  unsigned short* xb  = alloc(NX * 2);
  unsigned short* Wqb = alloc(NW * 2);
  unsigned short* Wkb = alloc(NW * 2);
  unsigned short* Wvb = alloc(NW * 2);
  unsigned short* Wob = alloc(NO * 2);
  unsigned short* Qb  = alloc(NQ * 2);
  unsigned short* Kb  = alloc(NQ * 2);
  unsigned short* Vbf = alloc(NQ * 2);
  unsigned short* Zb  = alloc(NX * 2);

  // fp32 -> bf16 conversions (float4 / ushort4 vectorized)
  cvt_f32_bf16_kernel<<<(int)((NX / 4 + 255) / 256), 256, 0, stream>>>(x,  xb,  (int)(NX / 4));
  cvt_f32_bf16_kernel<<<(int)((NW / 4 + 255) / 256), 256, 0, stream>>>(Wq, Wqb, (int)(NW / 4));
  cvt_f32_bf16_kernel<<<(int)((NW / 4 + 255) / 256), 256, 0, stream>>>(Wk, Wkb, (int)(NW / 4));
  cvt_f32_bf16_kernel<<<(int)((NW / 4 + 255) / 256), 256, 0, stream>>>(Wv, Wvb, (int)(NW / 4));
  cvt_f32_bf16_kernel<<<(int)((NO / 4 + 255) / 256), 256, 0, stream>>>(Wo, Wob, (int)(NO / 4));

  // Q/K/V projections: 128 M-tiles x 16 heads x {q,k,v}
  qkv_proj_kernel<<<dim3(128, N_HEADS, 3), 128, 0, stream>>>(
      xb, Wqb, Wkb, Wvb, bq, bk, bv, Qb, Kb, Vbf);

  // flash attention: 32 q-tiles x 16 heads x 4 batches
  attn_kernel<<<dim3(SEQ / 64, N_HEADS, BATCH), 128, 0, stream>>>(Qb, Kb, Vbf, Zb);

  // output projection: 128 M-tiles x 16 N-tiles
  oproj_kernel<<<dim3(128, D_MODEL / 64), 128, 0, stream>>>(Zb, Wob, bo, out);
}
